// MultiHeadAttention_317827579926
// MI455X (gfx1250) — matrix operand order repaired and checked by reading, not compiled, not run on hardware
//
#include <hip/hip_runtime.h>

#define B_    2
#define S_    2048
#define HID_  1024
#define H_    16
#define DK_   64
#define NTOT_ 1024

typedef __attribute__((ext_vector_type(16))) __bf16 v16bf;
typedef __attribute__((ext_vector_type(8)))  float  v8f;

union FragAB { v16bf v; unsigned int u[8]; };

__device__ __forceinline__ unsigned short f2bf(float f) {
    unsigned int u = __float_as_uint(f);
    u += 0x7FFFu + ((u >> 16) & 1u);          // round-to-nearest-even
    return (unsigned short)(u >> 16);
}
__device__ __forceinline__ unsigned int pack2bf(float a, float b) {
    return (unsigned int)f2bf(a) | ((unsigned int)f2bf(b) << 16);
}

// ---------------------------------------------------------------------------
// Generic GEMM: Out = (X @ W + bias) * scale
// X: M x K (fp32 or bf16), W: K x N fp32 row-major.
// OUT_MODE 0: bf16 output in [B,H,S,DK] layout (row=b*S+s, col=h*64+d)
// OUT_MODE 1: fp32 output row-major M x N (non-temporal: never re-read)
// Block = 256 threads (8 waves), block tile 128x128, BK=32.
// Wave tile 32x64 -> 2x4 WMMA accumulators of 16x16.
// ---------------------------------------------------------------------------
template<bool A_BF16, int OUT_MODE>
__global__ __launch_bounds__(256) void gemm_bias_kernel(
    const void* __restrict__ Xv, const float* __restrict__ W,
    const float* __restrict__ bias, void* __restrict__ Outv,
    int M, int N, int K, float scale)
{
    __shared__ unsigned short As[128][36];   // padded: 72B row stride
    __shared__ unsigned short Bs[128][36];   // W tile transposed: Bs[n][k]

    const int tid  = threadIdx.x;
    const int lane = tid & 31, wave = tid >> 5;
    const int ln   = lane & 15, hi = lane >> 4;
    const int waveM = wave & 3, waveN = wave >> 2;
    const int mBase = blockIdx.y * 128;
    const int nBase = blockIdx.x * 128;

    v8f zero = {};
    v8f acc[2][4];
    #pragma unroll
    for (int i = 0; i < 2; i++)
        #pragma unroll
        for (int j = 0; j < 4; j++) acc[i][j] = zero;

    for (int k0 = 0; k0 < K; k0 += 32) {
        // --- X tile 128x32 -> As (bf16) ---
        #pragma unroll
        for (int i = 0; i < 4; i++) {
            int p   = tid + i * 256;       // 1024 groups of 4 elements
            int row = p >> 3;
            int c4  = (p & 7) << 2;
            unsigned int lo, hi2;
            if constexpr (A_BF16) {
                const unsigned short* X = (const unsigned short*)Xv;
                const unsigned int* src =
                    (const unsigned int*)(X + (size_t)(mBase + row) * K + k0 + c4);
                lo = src[0]; hi2 = src[1];
            } else {
                const float* X = (const float*)Xv;
                float4 vv = *(const float4*)(X + (size_t)(mBase + row) * K + k0 + c4);
                lo = pack2bf(vv.x, vv.y); hi2 = pack2bf(vv.z, vv.w);
            }
            *(unsigned int*)&As[row][c4]     = lo;
            *(unsigned int*)&As[row][c4 + 2] = hi2;
        }
        // --- W tile 32x128 -> Bs transposed (bf16) ---
        #pragma unroll
        for (int i = 0; i < 4; i++) {
            int p    = tid + i * 256;
            int wrow = p >> 5;
            int c4   = (p & 31) << 2;
            float4 vv = *(const float4*)(W + (size_t)(k0 + wrow) * N + nBase + c4);
            Bs[c4 + 0][wrow] = f2bf(vv.x);
            Bs[c4 + 1][wrow] = f2bf(vv.y);
            Bs[c4 + 2][wrow] = f2bf(vv.z);
            Bs[c4 + 3][wrow] = f2bf(vv.w);
        }
        __syncthreads();

        // A fragments: lane = M row; pairs at k = (e/8)*16 + hi*8 + (e%8)
        FragAB a[2];
        #pragma unroll
        for (int i = 0; i < 2; i++) {
            int row = waveM * 32 + i * 16 + ln;
            #pragma unroll
            for (int p = 0; p < 8; p++) {
                int kk = ((p >> 2) << 4) + hi * 8 + ((p & 3) << 1);
                a[i].u[p] = *(const unsigned int*)&As[row][kk];
            }
        }
        #pragma unroll
        for (int j = 0; j < 4; j++) {
            // B fragment: lane = N col; k = (e/8)*16 + hi*8 + (e%8)
            FragAB bfrag;
            int n = waveN * 64 + j * 16 + ln;
            #pragma unroll
            for (int p = 0; p < 8; p++) {
                int kk = ((p >> 2) << 4) + hi * 8 + ((p & 3) << 1);
                bfrag.u[p] = *(const unsigned int*)&Bs[n][kk];
            }
            #pragma unroll
            for (int i = 0; i < 2; i++) {
                acc[i][j] = __builtin_amdgcn_wmma_f32_16x16x32_bf16(
                    false, a[i].v, false, bfrag.v, (short)0, acc[i][j], false, false);
            }
        }
        __syncthreads();
    }

    // Epilogue: C layout -> row = hi*8 + r, col = ln within 16x16 tile
    #pragma unroll
    for (int i = 0; i < 2; i++) {
        #pragma unroll
        for (int j = 0; j < 4; j++) {
            int col = nBase + waveN * 64 + j * 16 + ln;
            float bcol = bias[col];
            #pragma unroll
            for (int r = 0; r < 8; r++) {
                int row = mBase + waveM * 32 + i * 16 + hi * 8 + r;
                float val = (acc[i][j][r] + bcol) * scale;
                if constexpr (OUT_MODE == 0) {
                    unsigned short* Out = (unsigned short*)Outv;
                    int b = row >> 11, s = row & (S_ - 1);
                    int h = col >> 6,  d = col & (DK_ - 1);
                    Out[(((size_t)(b * H_ + h)) * S_ + s) * DK_ + d] = f2bf(val);
                } else {
                    float* Out = (float*)Outv;
                    // final result: written once, never re-read on device
                    __builtin_nontemporal_store(val, &Out[(size_t)row * N + col]);
                }
            }
        }
    }
}

// ---------------------------------------------------------------------------
// Flash attention: each wave owns 16 query rows, streams keys in chunks of 32.
// Block = 256 threads (8 waves) -> 128 query rows per block.
// Grid: (S/128, B*H).
// attn_bias (512 MB, read exactly once) is loaded non-temporal so the
// streaming read does not evict K/V (L2-resident, re-read 16x) from the
// 192 MB L2.
// ---------------------------------------------------------------------------
__global__ __launch_bounds__(256) void attn_kernel(
    const unsigned short* __restrict__ qh,   // [B,H,S,DK] bf16, pre-scaled
    const unsigned short* __restrict__ kh,   // [B,H,S,DK] bf16
    const unsigned short* __restrict__ vh,   // [B,H,S,DK] bf16
    const float* __restrict__ attn_bias,     // [B,H,S,S] f32
    const int* __restrict__ mask,            // [B,S] i32
    unsigned short* __restrict__ xout)       // [B,S,H*DK] bf16
{
    __shared__ unsigned short Ks[32][68];     // key-major, padded
    __shared__ unsigned short Vt[64][36];     // d-major (transposed), padded
    __shared__ unsigned short Ps[8][16][36];  // per-wave P scratch, padded

    const int tid  = threadIdx.x;
    const int lane = tid & 31, wave = tid >> 5;
    const int ln   = lane & 15, hi = lane >> 4;
    const int bh   = blockIdx.y;
    const int b    = bh >> 4, h = bh & 15;
    const int qBase = blockIdx.x * 128 + wave * 16;

    const size_t headBase = (size_t)bh * S_ * DK_;

    // Q A-fragments (16 rows x 64 d), kept in registers for all key chunks
    FragAB qA[2];
    {
        const unsigned short* qrow = qh + headBase + (size_t)(qBase + ln) * DK_;
        #pragma unroll
        for (int f = 0; f < 2; f++)
            #pragma unroll
            for (int p = 0; p < 8; p++) {
                int kk = f * 32 + ((p >> 2) << 4) + hi * 8 + ((p & 3) << 1);
                qA[f].u[p] = *(const unsigned int*)(qrow + kk);
            }
    }

    v8f zero = {};
    v8f accO[4];
    #pragma unroll
    for (int j = 0; j < 4; j++) accO[j] = zero;
    float m_r[8], l_r[8];
    #pragma unroll
    for (int r = 0; r < 8; r++) { m_r[r] = -3.0e38f; l_r[r] = 0.0f; }

    for (int kc = 0; kc < S_; kc += 32) {
        // stage keys kc..kc+15 of K chunk, 16x64 (row-major)
        #pragma unroll
        for (int i = 0; i < 2; i++) {
            int p   = tid + i * 256;           // 512 dword positions
            int key = p >> 5;
            int du  = (p & 31) << 1;
            unsigned int vv = *(const unsigned int*)
                (kh + headBase + (size_t)(kc + key) * DK_ + du);
            *(unsigned int*)&Ks[key][du] = vv;
        }
        // stage V chunk transposed 64x32
        #pragma unroll
        for (int i = 0; i < 8; i++) {
            int p   = tid + i * 256;           // 2048 elements
            int key = p >> 6;
            int d   = p & 63;
            Vt[d][key] = vh[headBase + (size_t)(kc + key) * DK_ + d];
        }
        __syncthreads();

        // scores S = Q @ K^T : two 16x16 tiles (keys kc..+15, kc+16..+31)
        v8f sc[2];
        #pragma unroll
        for (int t = 0; t < 2; t++) {
            v8f c = zero;
            #pragma unroll
            for (int f = 0; f < 2; f++) {       // contraction over d in chunks of 32
                FragAB bfrag;
                #pragma unroll
                for (int p = 0; p < 8; p++) {
                    int kk = f * 32 + ((p >> 2) << 4) + hi * 8 + ((p & 3) << 1);
                    bfrag.u[p] = *(const unsigned int*)&Ks[t * 16 + ln][kk];
                }
                c = __builtin_amdgcn_wmma_f32_16x16x32_bf16(
                    false, qA[f].v, false, bfrag.v, (short)0, c, false, false);
            }
            sc[t] = c;
        }

        // key mask + bias (reference: where(mask==0, NEG, s) + bias)
        // bias is a one-shot 512 MB stream -> non-temporal loads
        #pragma unroll
        for (int t = 0; t < 2; t++) {
            int key = kc + t * 16 + ln;
            int km  = mask[b * S_ + key];
            const float* brow = attn_bias + ((size_t)bh * S_ + (qBase + hi * 8)) * S_ + key;
            #pragma unroll
            for (int r = 0; r < 8; r++) {
                float s  = km ? sc[t][r] : -1.0e9f;
                float bv = __builtin_nontemporal_load(&brow[(size_t)r * S_]);
                sc[t][r] = s + bv;
            }
        }

        // online softmax (row lives in 16 lanes of a half-wave)
        float alpha[8];
        #pragma unroll
        for (int r = 0; r < 8; r++) {
            float cm = fmaxf(sc[0][r], sc[1][r]);
            #pragma unroll
            for (int off = 1; off < 16; off <<= 1)
                cm = fmaxf(cm, __shfl_xor(cm, off, 32));
            float mn = fmaxf(m_r[r], cm);
            alpha[r] = __expf(m_r[r] - mn);
            m_r[r]   = mn;
            float p0 = __expf(sc[0][r] - mn);
            float p1 = __expf(sc[1][r] - mn);
            sc[0][r] = p0; sc[1][r] = p1;
            float rs = p0 + p1;
            #pragma unroll
            for (int off = 1; off < 16; off <<= 1)
                rs += __shfl_xor(rs, off, 32);
            l_r[r] = l_r[r] * alpha[r] + rs;
        }
        #pragma unroll
        for (int j = 0; j < 4; j++)
            #pragma unroll
            for (int r = 0; r < 8; r++)
                accO[j][r] *= alpha[r];

        // P: C-layout -> LDS (per-wave, same-wave LDS ops are in-order)
        #pragma unroll
        for (int t = 0; t < 2; t++)
            #pragma unroll
            for (int r = 0; r < 8; r++)
                Ps[wave][hi * 8 + r][t * 16 + ln] = f2bf(sc[t][r]);

        // read P back as A-fragment (16 rows x 32 keys)
        FragAB pa;
        #pragma unroll
        for (int p = 0; p < 8; p++) {
            int kk = ((p >> 2) << 4) + hi * 8 + ((p & 3) << 1);
            pa.u[p] = *(const unsigned int*)&Ps[wave][ln][kk];
        }

        // O += P @ V
        #pragma unroll
        for (int j = 0; j < 4; j++) {
            FragAB bfrag;
            #pragma unroll
            for (int p = 0; p < 8; p++) {
                int kk = ((p >> 2) << 4) + hi * 8 + ((p & 3) << 1);
                bfrag.u[p] = *(const unsigned int*)&Vt[j * 16 + ln][kk];
            }
            accO[j] = __builtin_amdgcn_wmma_f32_16x16x32_bf16(
                false, pa.v, false, bfrag.v, (short)0, accO[j], false, false);
        }
        __syncthreads();
    }

    // finalize: divide by l, apply query mask, write bf16 x in [B,S,H*DK]
    #pragma unroll
    for (int r = 0; r < 8; r++) {
        int qrow = qBase + hi * 8 + r;
        float f = (1.0f / l_r[r]) * (mask[b * S_ + qrow] ? 1.0f : 0.0f);
        #pragma unroll
        for (int j = 0; j < 4; j++) {
            int d = j * 16 + ln;
            xout[((size_t)(b * S_) + qrow) * NTOT_ + h * DK_ + d] =
                f2bf(accO[j][r] * f);
        }
    }
}

// ---------------------------------------------------------------------------
extern "C" void kernel_launch(void* const* d_in, const int* in_sizes, int n_in,
                              void* d_out, int out_size, void* d_ws, size_t ws_size,
                              hipStream_t stream)
{
    const float* q  = (const float*)d_in[0];
    const float* k  = (const float*)d_in[1];
    const float* v  = (const float*)d_in[2];
    const float* attn_bias = (const float*)d_in[3];
    const int*   mask = (const int*)d_in[4];
    const float* Wq = (const float*)d_in[5];
    const float* bq = (const float*)d_in[6];
    const float* Wk = (const float*)d_in[7];
    const float* bk = (const float*)d_in[8];
    const float* Wv = (const float*)d_in[9];
    const float* bv = (const float*)d_in[10];
    const float* Wo = (const float*)d_in[11];
    const float* bo = (const float*)d_in[12];
    float* out = (float*)d_out;

    const size_t HEADELEMS = (size_t)B_ * H_ * S_ * DK_;   // 4M elems
    unsigned short* qh = (unsigned short*)d_ws;            // 8 MiB each
    unsigned short* kh = qh + HEADELEMS;
    unsigned short* vh = kh + HEADELEMS;
    unsigned short* xb = vh + HEADELEMS;

    const int M = B_ * S_, N = NTOT_, K = HID_;
    dim3 blk(256);
    dim3 gg(N / 128, M / 128);
    const float SCALE = 0.125f;   // DK^-0.5

    gemm_bias_kernel<false, 0><<<gg, blk, 0, stream>>>(q, Wq, bq, qh, M, N, K, SCALE);
    gemm_bias_kernel<false, 0><<<gg, blk, 0, stream>>>(k, Wk, bk, kh, M, N, K, 1.0f);
    gemm_bias_kernel<false, 0><<<gg, blk, 0, stream>>>(v, Wv, bv, vh, M, N, K, 1.0f);

    dim3 ga(S_ / 128, B_ * H_);
    attn_kernel<<<ga, blk, 0, stream>>>(qh, kh, vh, attn_bias, mask, xb);

    gemm_bias_kernel<true, 1><<<gg, blk, 0, stream>>>(xb, Wo, bo, out, M, N, K, 1.0f);
}
